// MoDMLPWrapper_59657095741546
// MI455X (gfx1250) — matrix operand order repaired and checked by reading, not compiled, not run on hardware
//
#include <hip/hip_runtime.h>
#include <math.h>

// ---------------- problem constants ----------------
constexpr int Bc    = 4;
constexpr int Lc    = 4096;
constexpr int Dc    = 1024;
constexpr int DFFc  = 4096;
constexpr int KEEPc = 2048;          // K = L * 0.5
constexpr int MTOT  = Bc * KEEPc;    // 8192 rows through the MLP

typedef unsigned short bhalf;        // raw bf16 storage

// ---------------- CDNA5 WMMA types ----------------
typedef __attribute__((ext_vector_type(16))) __bf16        v16bf;
typedef __attribute__((ext_vector_type(8)))  float         v8f;
typedef __attribute__((ext_vector_type(4)))  unsigned int  u32x4;

union FragCvt { u32x4 u[2]; v16bf v; };   // 32B: one 16-bf16 WMMA operand

// ---------------- helpers ----------------
__device__ __forceinline__ bhalf f2bf(float f) {
  unsigned int u = __builtin_bit_cast(unsigned int, f);
  u += 0x7FFFu + ((u >> 16) & 1u);        // round-to-nearest-even
  return (bhalf)(u >> 16);
}

__device__ __forceinline__ float gelu_tanh(float x) {
  // jax.nn.gelu default (approximate=True)
  const float c0 = 0.7978845608028654f;   // sqrt(2/pi)
  const float c1 = 0.044715f;
  float t = c0 * (x + c1 * x * x * x);
  return 0.5f * x * (1.0f + tanhf(t));
}

__device__ __forceinline__ float wave_sum_f(float v) {
  for (int o = 16; o; o >>= 1) v += __shfl_xor(v, o, 32);
  return v;
}
__device__ __forceinline__ int wave_sum_i(int v) {
  for (int o = 16; o; o >>= 1) v += __shfl_xor(v, o, 32);
  return v;
}

// ---------------- 1) router scores: one wave per token ----------------
__global__ __launch_bounds__(256)
void router_kernel(const float* __restrict__ x, const float* __restrict__ wr,
                   const float* __restrict__ br, float* __restrict__ scores) {
  const int t = threadIdx.x, lane = t & 31, wid = t >> 5;
  const int tok = blockIdx.x * 8 + wid;            // [0, B*L)
  const float* xr = x + (size_t)tok * Dc;
  float s = 0.f;
  for (int d = lane; d < Dc; d += 32) s += xr[d] * wr[d];
  s = wave_sum_f(s);
  if (lane == 0) scores[tok] = s + br[0];
}

// ---------------- 2) stable rank -> keep flag (top-K, ties by index) ----------------
__global__ __launch_bounds__(256)
void rank_kernel(const float* __restrict__ scores, int* __restrict__ keep) {
  const int t = threadIdx.x, lane = t & 31, wid = t >> 5;
  const int g = blockIdx.x * 8 + wid;              // global token id
  const int b = g / Lc, i = g - b * Lc;
  const float si = scores[g];
  const float* sb = scores + (size_t)b * Lc;
  int cnt = 0;
  for (int j = lane; j < Lc; j += 32) {
    float sj = sb[j];
    cnt += (sj > si) || (sj == si && j < i);       // # strictly-better tokens
  }
  cnt = wave_sum_i(cnt);
  if (lane == 0) keep[g] = (cnt < KEEPc) ? 1 : 0;
}

// ---------------- 3) in-order compaction -> sorted index list ----------------
__global__ __launch_bounds__(256)
void compact_kernel(const int* __restrict__ keep, int* __restrict__ idxl) {
  const int b = blockIdx.x, t = threadIdx.x;
  __shared__ int sc[256];
  int base = 0;
  for (int c = 0; c < Lc / 256; ++c) {
    const int i = c * 256 + t;
    const int v = keep[b * Lc + i];
    sc[t] = v; __syncthreads();
    for (int off = 1; off < 256; off <<= 1) {      // Hillis-Steele inclusive scan
      int add = (t >= off) ? sc[t - off] : 0;
      __syncthreads();
      sc[t] += add;
      __syncthreads();
    }
    const int excl = sc[t] - v;
    const int tot  = sc[255];
    if (v) idxl[b * KEEPc + base + excl] = i;
    base += tot;
    __syncthreads();
  }
}

// ---------------- 4) gather selected rows + cast to bf16 ----------------
__global__ __launch_bounds__(256)
void gather_cast_kernel(const float* __restrict__ x, const int* __restrict__ idxl,
                        bhalf* __restrict__ Xg) {
  const int row = blockIdx.x, t = threadIdx.x;     // row in [0, MTOT)
  const int b = row / KEEPc;
  const int tok = idxl[row];
  const float4 v = ((const float4*)(x + ((size_t)(b * Lc + tok)) * Dc))[t];
  unsigned int p0 = (unsigned int)f2bf(v.x) | ((unsigned int)f2bf(v.y) << 16);
  unsigned int p1 = (unsigned int)f2bf(v.z) | ((unsigned int)f2bf(v.w) << 16);
  ((uint2*)(Xg + (size_t)row * Dc))[t] = make_uint2(p0, p1);
}

// ---------------- transpose + cast: W[K][N] f32 -> Wt[N][K] bf16 ----------------
__global__ __launch_bounds__(256)
void transpose_cast_kernel(const float* __restrict__ W, bhalf* __restrict__ Wt,
                           int Kdim, int Ndim) {
  __shared__ float sh[32][33];
  const int t  = threadIdx.x;
  const int n0 = blockIdx.x * 32;
  const int k0 = blockIdx.y * 32;
  const int r  = t >> 3;          // 0..31
  const int c4 = (t & 7) << 2;    // 0,4,...,28

  const float4 v = *(const float4*)(W + (size_t)(k0 + r) * Ndim + n0 + c4);
  sh[r][c4 + 0] = v.x; sh[r][c4 + 1] = v.y; sh[r][c4 + 2] = v.z; sh[r][c4 + 3] = v.w;
  __syncthreads();

  const float a = sh[c4 + 0][r], b = sh[c4 + 1][r];
  const float c = sh[c4 + 2][r], d = sh[c4 + 3][r];
  unsigned int p0 = (unsigned int)f2bf(a) | ((unsigned int)f2bf(b) << 16);
  unsigned int p1 = (unsigned int)f2bf(c) | ((unsigned int)f2bf(d) << 16);
  *(uint2*)(Wt + (size_t)(n0 + r) * Kdim + k0 + c4) = make_uint2(p0, p1);
}

// ---------------- zero output ----------------
__global__ __launch_bounds__(256)
void zero_kernel(float* __restrict__ out, int n4) {
  int i = blockIdx.x * blockDim.x + threadIdx.x;
  const int stride = gridDim.x * blockDim.x;
  const float4 z = {0.f, 0.f, 0.f, 0.f};
  for (; i < n4; i += stride) ((float4*)out)[i] = z;
}

// ---------------- WMMA bf16 GEMM: C = A[M,K] * Bt[N,K]^T (+bias, +epilogue) ------------
// Block: 256 threads = 8 waves (4x2), block tile 128x128, K-step 32.
// Each wave: 32x64 region = 2x4 v_wmma_f32_16x16x32_bf16 tiles.
// LDS ping-pong double buffering: one barrier per k-step; global loads for the
// next tile issue before the current tile's WMMAs; prefetch runs 2 tiles ahead.
template<int NDIM, int KDIM, bool GELU_BF16_OUT>
__global__ __launch_bounds__(256)
void gemm_bf16_kernel(const bhalf* __restrict__ A, const bhalf* __restrict__ Bt,
                      const float* __restrict__ bias,
                      bhalf* __restrict__ outBf, float* __restrict__ outF,
                      const int* __restrict__ idxl) {
  constexpr int BM = 128, BN = 128, BK = 32, LDP = 40;  // LDP*2B = 80B stride (16B aligned)
  __shared__ __align__(16) bhalf lsA[2][BM][LDP];       // [buf][m][k]
  __shared__ __align__(16) bhalf lsB[2][BN][LDP];       // [buf][n][k]

  const int t = threadIdx.x;
  const int lane = t & 31, w = t >> 5;
  const int wm = w >> 1, wn = w & 1;                    // wave grid 4x2
  const int lm = lane & 15, lh = lane >> 4;             // lane row/col, half select
  const int m0 = blockIdx.y * BM;
  const int n0 = blockIdx.x * BN;

  const v8f z8 = {0.f, 0.f, 0.f, 0.f, 0.f, 0.f, 0.f, 0.f};
  v8f acc[2][4];
  #pragma unroll
  for (int i = 0; i < 2; ++i)
    #pragma unroll
    for (int j = 0; j < 4; ++j) acc[i][j] = z8;

  const int srow = t >> 1, shalf = t & 1;               // staging: 2 thr/row, 32B each
  const bhalf* gA = A  + (size_t)(m0 + srow) * KDIM + shalf * 16;
  const bhalf* gB = Bt + (size_t)(n0 + srow) * KDIM + shalf * 16;

  // ---- prologue: stage tile 0 into buffer 0 ----
  {
    const u32x4* pa = (const u32x4*)gA;
    const u32x4* pb = (const u32x4*)gB;
    u32x4 a0 = pa[0], a1 = pa[1], b0 = pb[0], b1 = pb[1];
    *(u32x4*)&lsA[0][srow][shalf * 16]     = a0;
    *(u32x4*)&lsA[0][srow][shalf * 16 + 8] = a1;
    *(u32x4*)&lsB[0][srow][shalf * 16]     = b0;
    *(u32x4*)&lsB[0][srow][shalf * 16 + 8] = b1;
  }
  __syncthreads();

  int buf = 0;
  for (int k0 = 0; k0 < KDIM; k0 += BK, buf ^= 1) {
    const bool more = (k0 + BK) < KDIM;

    // ---- issue next tile's global loads early (into registers) ----
    u32x4 ra0, ra1, rb0, rb1;
    if (more) {
      const u32x4* pa = (const u32x4*)(gA + k0 + BK);
      const u32x4* pb = (const u32x4*)(gB + k0 + BK);
      ra0 = pa[0]; ra1 = pa[1]; rb0 = pb[0]; rb1 = pb[1];
      if (k0 + 2 * BK < KDIM) {                         // L2 prefetch, 2 tiles ahead
        __builtin_prefetch(gA + k0 + 2 * BK, 0, 1);
        __builtin_prefetch(gB + k0 + 2 * BK, 0, 1);
      }
    }

    // ---- fragments per ISA 16-bit layouts, from current buffer ----
    //  A lane(l): m = l%16; e0..7 -> k = e + 8*(l/16); e8..15 -> k = 16 + (e-8) + 8*(l/16)
    //  B lane(l): n = l%16; e0..7 -> k = e + 8*(l/16); e8..15 -> k = 16 + (e-8) + 8*(l/16)
    FragCvt af[2], bf[4];
    #pragma unroll
    for (int i = 0; i < 2; ++i) {
      const int r = wm * 32 + i * 16 + lm;
      af[i].u[0] = *(const u32x4*)&lsA[buf][r][8 * lh];
      af[i].u[1] = *(const u32x4*)&lsA[buf][r][16 + 8 * lh];
    }
    #pragma unroll
    for (int j = 0; j < 4; ++j) {
      const int n = wn * 64 + j * 16 + lm;
      bf[j].u[0] = *(const u32x4*)&lsB[buf][n][8 * lh];
      bf[j].u[1] = *(const u32x4*)&lsB[buf][n][16 + 8 * lh];
    }
    #pragma unroll
    for (int i = 0; i < 2; ++i)
      #pragma unroll
      for (int j = 0; j < 4; ++j)
        acc[i][j] = __builtin_amdgcn_wmma_f32_16x16x32_bf16(
            false, af[i].v, false, bf[j].v, (short)0, acc[i][j], false, false);

    // ---- write next tile into the other buffer ----
    if (more) {
      const int nb = buf ^ 1;
      *(u32x4*)&lsA[nb][srow][shalf * 16]     = ra0;
      *(u32x4*)&lsA[nb][srow][shalf * 16 + 8] = ra1;
      *(u32x4*)&lsB[nb][srow][shalf * 16]     = rb0;
      *(u32x4*)&lsB[nb][srow][shalf * 16 + 8] = rb1;
      __syncthreads();   // single barrier per k-step (read-cur / write-next handoff)
    }
  }

  // ---- epilogue: D VGPR r -> m = r + 8*lh, n = lm ----
  #pragma unroll
  for (int i = 0; i < 2; ++i) {
    #pragma unroll
    for (int r = 0; r < 8; ++r) {
      const int row = m0 + wm * 32 + i * 16 + r + 8 * lh;
      size_t obase;
      if constexpr (GELU_BF16_OUT) {
        obase = (size_t)row * NDIM;
      } else {
        const int b = row / KEEPc;
        const int tok = idxl[row];                     // fused scatter
        obase = ((size_t)(b * Lc + tok)) * NDIM;
      }
      #pragma unroll
      for (int j = 0; j < 4; ++j) {
        const int col = n0 + wn * 64 + j * 16 + lm;
        float v = acc[i][j][r] + bias[col];
        if constexpr (GELU_BF16_OUT) outBf[obase + col] = f2bf(gelu_tanh(v));
        else                         outF[obase + col] = v;
      }
    }
  }
}

// ---------------- launch ----------------
extern "C" void kernel_launch(void* const* d_in, const int* in_sizes, int n_in,
                              void* d_out, int out_size, void* d_ws, size_t ws_size,
                              hipStream_t stream) {
  const float* x  = (const float*)d_in[0];
  const float* W1 = (const float*)d_in[1];
  const float* b1 = (const float*)d_in[2];
  const float* W2 = (const float*)d_in[3];
  const float* b2 = (const float*)d_in[4];
  const float* wr = (const float*)d_in[5];
  const float* br = (const float*)d_in[6];
  float* out = (float*)d_out;

  // workspace carve (~96.2 MB)
  char* p = (char*)d_ws;
  float* scores = (float*)p;  p += (size_t)Bc * Lc * 4;
  int*   keep   = (int*)p;    p += (size_t)Bc * Lc * 4;
  int*   idxl   = (int*)p;    p += (size_t)Bc * KEEPc * 4;
  bhalf* W1t    = (bhalf*)p;  p += (size_t)Dc * DFFc * 2;   // [DFF][D]  = W1^T bf16
  bhalf* W2t    = (bhalf*)p;  p += (size_t)DFFc * Dc * 2;   // [D][DFF]  = W2^T bf16
  bhalf* Xg     = (bhalf*)p;  p += (size_t)MTOT * Dc * 2;   // [MTOT][D]
  bhalf* H      = (bhalf*)p;  p += (size_t)MTOT * DFFc * 2; // [MTOT][DFF]

  router_kernel <<<Bc * Lc / 8, 256, 0, stream>>>(x, wr, br, scores);
  rank_kernel   <<<Bc * Lc / 8, 256, 0, stream>>>(scores, keep);
  compact_kernel<<<Bc,          256, 0, stream>>>(keep, idxl);
  gather_cast_kernel<<<MTOT,    256, 0, stream>>>(x, idxl, Xg);

  // W1 [D][DFF] -> W1t [DFF][D];  W2 [DFF][D] -> W2t [D][DFF]
  transpose_cast_kernel<<<dim3(DFFc / 32, Dc / 32), 256, 0, stream>>>(W1, W1t, Dc, DFFc);
  transpose_cast_kernel<<<dim3(Dc / 32, DFFc / 32), 256, 0, stream>>>(W2, W2t, DFFc, Dc);

  zero_kernel<<<4096, 256, 0, stream>>>(out, Bc * Lc * Dc / 4);

  // GEMM1: [8192 x 1024] x W1 -> GELU -> bf16 H
  gemm_bf16_kernel<DFFc, Dc, true>
      <<<dim3(DFFc / 128, MTOT / 128), 256, 0, stream>>>(Xg, W1t, b1, H, nullptr, nullptr);
  // GEMM2: [8192 x 4096] x W2 -> +b2 -> scattered f32 out
  gemm_bf16_kernel<Dc, DFFc, false>
      <<<dim3(Dc / 128, MTOT / 128), 256, 0, stream>>>(H, W2t, b2, nullptr, out, idxl);
}
